// HAN_33698313404426
// MI455X (gfx1250) — matrix operand order repaired and checked by reading, not compiled, not run on hardware
//
#include <hip/hip_runtime.h>
#include <math.h>

#define N_NODES 20000
#define N_EDGES 320000
#define IN_DIM  768
#define HID     64
#define HEADS   8
#define HD      512
#define NCLS    350

// -------- CDNA5 WMMA types --------
typedef __attribute__((ext_vector_type(16))) __bf16 v16bf;
typedef __attribute__((ext_vector_type(8)))  float  v8f;

union FragU { uint4 u4[2]; v16bf v; };

__device__ __forceinline__ unsigned short f2bf(float x) {
  unsigned u = __float_as_uint(x);
  u += 0x7FFFu + ((u >> 16) & 1u);           // round-to-nearest-even
  return (unsigned short)(u >> 16);
}
// ordered-uint encoding so unsigned atomicMax == float max
__device__ __forceinline__ unsigned f2o(float f) {
  unsigned u = __float_as_uint(f);
  return (u & 0x80000000u) ? ~u : (u | 0x80000000u);
}
__device__ __forceinline__ float o2f(unsigned u) {
  u = (u & 0x80000000u) ? (u & 0x7FFFFFFFu) : ~u;
  return __uint_as_float(u);
}
__device__ __forceinline__ float elu1(float v) { return v > 0.f ? v : (__expf(v) - 1.f); }
__device__ __forceinline__ float lrelu(float v) { return v > 0.f ? v : 0.2f * v; }
#define BN_SCALE 0.9999950000374997f   /* 1/sqrt(1+1e-5) */

// =====================================================================
// WMMA bf16 GEMM: C[M,N] = A[M,K] * B[K,N] (+ bias[N]); row-major f32.
// Requires K%32==0, N%128==0. Block 256 thr = 8 waves (2x4).
// Block tile 128x128, wave tile 64x32, double-buffered LDS staging.
// =====================================================================
#define BM 128
#define BN 128
#define BK 32
#define AP 40              // LDS pitch in halves (padded: 80B rows, conflict-free)
#define ASZ (BM * AP)      // 5120 halves per buffer
#define BSZ (BN * AP)      // 5120 halves per buffer

__global__ __launch_bounds__(256) void gemm_bf16_wmma(
    const float* __restrict__ A, const float* __restrict__ B,
    const float* __restrict__ bias, float* __restrict__ C,
    int M, int N, int K)
{
  __shared__ unsigned short As[2 * ASZ];  // [buf][row][k]
  __shared__ unsigned short Bs[2 * BSZ];  // [buf][n][k]  (transposed stage)

  const int tid  = threadIdx.x;
  const int lane = tid & 31;
  const int wave = tid >> 5;
  const int wm = wave & 1;        // 0..1 -> 64-row M sub-block
  const int wn = wave >> 1;       // 0..3 -> 32-col N sub-block
  const int m0 = blockIdx.y * BM;
  const int n0 = blockIdx.x * BN;
  const bool full = (m0 + BM) <= M;

  v8f acc[4][2] = {};             // 64x32 wave tile

  // A staging: 32 rows/pass, 8 thr * float4 per row; 4 passes
  const int ar = tid >> 3;
  const int ac = (tid & 7) * 4;
  // B staging: 8 k-rows/pass, 32 thr * float4 per row; 4 passes
  const int bkr = tid >> 5;
  const int bc  = (tid & 31) * 4;
  const int alow = lane & 15;
  const int asel = (lane < 16) ? 0 : 8;    // A frag K split per ISA layout
  const int bsel = (lane < 16) ? 0 : 8;    // B frag K split per ISA layout

  // ---- stage one K-tile (global f32 -> bf16 LDS) ----
  auto stageA = [&](int k0, int buf) {
    unsigned short* dstBase = &As[buf * ASZ];
    if (full) {
#pragma unroll
      for (int i = 0; i < 4; ++i) {
        int r = ar + 32 * i;
        float4 v = *(const float4*)&A[(size_t)(m0 + r) * K + k0 + ac];
        unsigned short* d = dstBase + r * AP + ac;
        d[0] = f2bf(v.x); d[1] = f2bf(v.y); d[2] = f2bf(v.z); d[3] = f2bf(v.w);
      }
    } else {
#pragma unroll
      for (int i = 0; i < 4; ++i) {
        int r = ar + 32 * i;
        float4 v = make_float4(0.f, 0.f, 0.f, 0.f);
        if (m0 + r < M) v = *(const float4*)&A[(size_t)(m0 + r) * K + k0 + ac];
        unsigned short* d = dstBase + r * AP + ac;
        d[0] = f2bf(v.x); d[1] = f2bf(v.y); d[2] = f2bf(v.z); d[3] = f2bf(v.w);
      }
    }
  };
  auto stageB = [&](int k0, int buf) {
    unsigned short* dstBase = &Bs[buf * BSZ];
#pragma unroll
    for (int i = 0; i < 4; ++i) {
      int kk = bkr + 8 * i;
      float4 v = *(const float4*)&B[(size_t)(k0 + kk) * N + n0 + bc];
      dstBase[(bc + 0) * AP + kk] = f2bf(v.x);
      dstBase[(bc + 1) * AP + kk] = f2bf(v.y);
      dstBase[(bc + 2) * AP + kk] = f2bf(v.z);
      dstBase[(bc + 3) * AP + kk] = f2bf(v.w);
    }
  };

  stageA(0, 0);
  stageB(0, 0);
  __syncthreads();

  int cur = 0;
  for (int k0 = 0; k0 < K; k0 += BK) {
    if (k0 + BK < K) {               // stage next tile into other buffer (overlaps WMMA)
      stageA(k0 + BK, cur ^ 1);
      stageB(k0 + BK, cur ^ 1);
    }
    const unsigned short* Ab = &As[cur * ASZ];
    const unsigned short* Bb = &Bs[cur * BSZ];

    // B fragments: column = lane&15, lane<16 -> K {0..7,16..23}, lane>=16 -> K {8..15,24..31}
    v16bf bfrag[2];
#pragma unroll
    for (int nt = 0; nt < 2; ++nt) {
      int col = wn * 32 + nt * 16 + alow;
      FragU u;
      u.u4[0] = *(const uint4*)&Bb[col * AP + bsel];
      u.u4[1] = *(const uint4*)&Bb[col * AP + bsel + 16];
      bfrag[nt] = u.v;
    }
    // A fragments: lane<16 holds K {0..7,16..23}, lane>=16 holds K {8..15,24..31}
#pragma unroll
    for (int mt = 0; mt < 4; ++mt) {
      int row = wm * 64 + mt * 16 + alow;
      FragU u;
      u.u4[0] = *(const uint4*)&Ab[row * AP + asel];
      u.u4[1] = *(const uint4*)&Ab[row * AP + asel + 16];
      v16bf afrag = u.v;
#pragma unroll
      for (int nt = 0; nt < 2; ++nt)
        acc[mt][nt] = __builtin_amdgcn_wmma_f32_16x16x32_bf16(
            false, afrag, false, bfrag[nt], (short)0, acc[mt][nt], false, false);
    }
    __syncthreads();
    cur ^= 1;
  }

  // C/D layout: VGPR j -> M=j (lanes 0-15) / M=j+8 (lanes 16-31), N = lane&15
#pragma unroll
  for (int mt = 0; mt < 4; ++mt) {
#pragma unroll
    for (int nt = 0; nt < 2; ++nt) {
      int col = n0 + wn * 32 + nt * 16 + (lane & 15);
      int rowBase = m0 + wm * 64 + mt * 16 + ((lane < 16) ? 0 : 8);
      float bv = bias ? bias[col] : 0.f;
      if (full) {
#pragma unroll
        for (int j = 0; j < 8; ++j)
          C[(size_t)(rowBase + j) * N + col] = acc[mt][nt][j] + bv;
      } else {
#pragma unroll
        for (int j = 0; j < 8; ++j) {
          int row = rowBase + j;
          if (row < M) C[(size_t)row * N + col] = acc[mt][nt][j] + bv;
        }
      }
    }
  }
}

// =====================================================================
// Fills
// =====================================================================
__global__ void k_fill_f32(float* p, float v, long n) {
  long i = blockIdx.x * (long)blockDim.x + threadIdx.x;
  if (i < n) p[i] = v;
}
__global__ void k_fill_u32(unsigned* p, unsigned v, long n) {
  long i = blockIdx.x * (long)blockDim.x + threadIdx.x;
  if (i < n) p[i] = v;
}

// =====================================================================
// GAT edge pipeline
// =====================================================================
__global__ void k_attn_logits(const float* __restrict__ f, const float* __restrict__ al,
                              const float* __restrict__ ar, float* el, float* er, long total) {
  long i = blockIdx.x * (long)blockDim.x + threadIdx.x;
  if (i >= total) return;
  int n = (int)(i >> 3), h = (int)(i & 7);
  const float* fr = f + (size_t)n * HD + h * HID;
  const float* pa = al + h * HID;
  const float* pb = ar + h * HID;
  float a = 0.f, b = 0.f;
#pragma unroll 4
  for (int d = 0; d < HID; ++d) { float fv = fr[d]; a += fv * pa[d]; b += fv * pb[d]; }
  el[i] = a; er[i] = b;
}

__global__ void k_edge_max(const float* __restrict__ el, const float* __restrict__ er,
                           const int* __restrict__ src, const int* __restrict__ dst,
                           float* ee, unsigned* mord, long total) {
  long i = blockIdx.x * (long)blockDim.x + threadIdx.x;
  if (i >= total) return;
  int e = (int)(i >> 3), h = (int)(i & 7);
  int s = src[e], d = dst[e];
  float v = lrelu(el[(size_t)s * 8 + h] + er[(size_t)d * 8 + h]);
  ee[i] = v;
  atomicMax(&mord[(size_t)d * 8 + h], f2o(v));
}

__global__ void k_edge_expsum(float* __restrict__ ee, const int* __restrict__ dst,
                              const unsigned* __restrict__ mord, float* ssum, long total) {
  long i = blockIdx.x * (long)blockDim.x + threadIdx.x;
  if (i >= total) return;
  int e = (int)(i >> 3), h = (int)(i & 7);
  int d = dst[e];
  float w = __expf(ee[i] - o2f(mord[(size_t)d * 8 + h]));
  ee[i] = w;
  atomicAdd(&ssum[(size_t)d * 8 + h], w);
}

__global__ void k_edge_agg(const float* __restrict__ ee, const float* __restrict__ ssum,
                           const float* __restrict__ f, const int* __restrict__ src,
                           const int* __restrict__ dst, float* out, long total) {
  long i = blockIdx.x * (long)blockDim.x + threadIdx.x;
  if (i >= total) return;
  int e = (int)(i >> 9);
  int r = (int)(i & 511);
  int h = r >> 6;
  int s = src[e], d = dst[e];
  float a = ee[(size_t)e * 8 + h] / (ssum[(size_t)d * 8 + h] + 1e-16f);
  atomicAdd(&out[(size_t)d * HD + r], f[(size_t)s * HD + r] * a);
}

__global__ void k_bias_elu(float* z, const float* __restrict__ b, long total) {
  long i = blockIdx.x * (long)blockDim.x + threadIdx.x;
  if (i >= total) return;
  z[i] = elu1(z[i] + b[i & (HD - 1)]);
}

// =====================================================================
// Semantic attention
// =====================================================================
__global__ __launch_bounds__(256) void k_sem_reduce(const float* __restrict__ t,
                                                    const float* __restrict__ b1,
                                                    const float* __restrict__ w2, float* wsum) {
  __shared__ float s[256];
  int n = blockIdx.x * blockDim.x + threadIdx.x;
  float acc = 0.f;
  if (n < N_NODES) {
    const float* r = t + (size_t)n * 128;
#pragma unroll 4
    for (int j = 0; j < 128; ++j) acc += tanhf(r[j] + b1[j]) * w2[j];
  }
  s[threadIdx.x] = acc; __syncthreads();
  for (int off = 128; off; off >>= 1) {
    if (threadIdx.x < off) s[threadIdx.x] += s[threadIdx.x + off];
    __syncthreads();
  }
  if (threadIdx.x == 0) atomicAdd(wsum, s[0]);
}

__global__ void k_beta(const float* wsum, float* beta) {
  float a = wsum[0] / (float)N_NODES, b = wsum[1] / (float)N_NODES;
  float m = fmaxf(a, b);
  float ea = __expf(a - m), eb = __expf(b - m);
  float inv = 1.f / (ea + eb);
  beta[0] = ea * inv; beta[1] = eb * inv;
}

__global__ void k_combine(const float* __restrict__ h3, const float* __restrict__ z0,
                          const float* __restrict__ z1, const float* __restrict__ beta,
                          const float* __restrict__ g, const float* __restrict__ b,
                          float* x, long total) {
  long i = blockIdx.x * (long)blockDim.x + threadIdx.x;
  if (i >= total) return;
  int d = (int)(i & (HD - 1));
  float v = h3[i] + beta[0] * z0[i] + beta[1] * z1[i];
  x[i] = elu1(g[d] * v * BN_SCALE + b[d]);
}

// =====================================================================
// Dots, softmax reductions, pooling
// =====================================================================
__global__ __launch_bounds__(256) void k_rowdot(const float* __restrict__ X,
                                                const float* __restrict__ w,
                                                const float* __restrict__ bias,
                                                float* out, int nrows, int D) {
  int wid = (int)((blockIdx.x * (long)blockDim.x + threadIdx.x) >> 5);
  int lane = threadIdx.x & 31;
  if (wid >= nrows) return;
  const float* row = X + (size_t)wid * D;
  float acc = 0.f;
  for (int i = lane; i < D; i += 32) acc += row[i] * w[i];
  for (int off = 16; off; off >>= 1) acc += __shfl_down(acc, off, 32);
  if (lane == 0) out[wid] = acc + (bias ? bias[0] : 0.f);
}

__global__ __launch_bounds__(256) void k_rowdot2(const float* __restrict__ Q,
                                                 const float* __restrict__ Km,
                                                 float* out, int nrows, int D, float scale) {
  int wid = (int)((blockIdx.x * (long)blockDim.x + threadIdx.x) >> 5);
  int lane = threadIdx.x & 31;
  if (wid >= nrows) return;
  const float* q = Q + (size_t)wid * D;
  const float* k = Km + (size_t)wid * D;
  float acc = 0.f;
  for (int i = lane; i < D; i += 32) acc += q[i] * k[i];
  for (int off = 16; off; off >>= 1) acc += __shfl_down(acc, off, 32);
  if (lane == 0) out[wid] = acc * scale;
}

__global__ __launch_bounds__(256) void k_reduce_max(const float* __restrict__ v, long n, unsigned* out) {
  __shared__ float s[256];
  float m = -3.4e38f;
  for (long j = blockIdx.x * (long)blockDim.x + threadIdx.x; j < n;
       j += (long)gridDim.x * blockDim.x) m = fmaxf(m, v[j]);
  s[threadIdx.x] = m; __syncthreads();
  for (int off = 128; off; off >>= 1) {
    if (threadIdx.x < off) s[threadIdx.x] = fmaxf(s[threadIdx.x], s[threadIdx.x + off]);
    __syncthreads();
  }
  if (threadIdx.x == 0) atomicMax(out, f2o(s[0]));
}

__global__ __launch_bounds__(256) void k_reduce_sumexp(const float* __restrict__ v, long n,
                                                       const unsigned* __restrict__ mord, float* out) {
  __shared__ float s[256];
  float mx = o2f(*mord);
  float acc = 0.f;
  for (long j = blockIdx.x * (long)blockDim.x + threadIdx.x; j < n;
       j += (long)gridDim.x * blockDim.x) acc += __expf(v[j] - mx);
  s[threadIdx.x] = acc; __syncthreads();
  for (int off = 128; off; off >>= 1) {
    if (threadIdx.x < off) s[threadIdx.x] += s[threadIdx.x + off];
    __syncthreads();
  }
  if (threadIdx.x == 0) atomicAdd(out, s[0]);
}

// out[d] = sum_n softmax(gv)[n] * X[n,d]
__global__ void k_wcolsum(const float* __restrict__ X, const float* __restrict__ gv,
                          const unsigned* __restrict__ mord, const float* __restrict__ sum,
                          float* out, int n, int D) {
  int d = blockIdx.x * blockDim.x + threadIdx.x;
  if (d >= D) return;
  float mx = o2f(*mord);
  float inv = 1.f / (*sum);
  float acc = 0.f;
  for (int i = 0; i < n; ++i) acc += __expf(gv[i] - mx) * inv * X[(size_t)i * D + d];
  out[d] = acc;
}

__global__ void k_vecmat(const float* __restrict__ v, const float* __restrict__ W,
                         const float* __restrict__ bias, float* out, int K, int Ncol) {
  int j = blockIdx.x * blockDim.x + threadIdx.x;
  if (j >= Ncol) return;
  float acc = bias ? bias[j] : 0.f;
  for (int k = 0; k < K; ++k) acc += v[k] * W[(size_t)k * Ncol + j];
  out[j] = acc;
}

// newT = bn3( vsum @ Wv )
__global__ void k_newT(const float* __restrict__ v, const float* __restrict__ W,
                       const float* __restrict__ g, const float* __restrict__ b,
                       float* out, int K, int Ncol) {
  int j = blockIdx.x * blockDim.x + threadIdx.x;
  if (j >= Ncol) return;
  float acc = 0.f;
  for (int k = 0; k < K; ++k) acc += v[k] * W[(size_t)k * Ncol + j];
  out[j] = g[j] * acc * BN_SCALE + b[j];
}

// =====================================================================
// LayerNorm (block per row): out = (ln(in + addb)) * g + b
// =====================================================================
__global__ __launch_bounds__(256) void k_layernorm(const float* __restrict__ in,
                                                   const float* __restrict__ addb,
                                                   const float* __restrict__ g,
                                                   const float* __restrict__ b,
                                                   float* out, int D) {
  __shared__ float s1[256], s2[256];
  int row = blockIdx.x, tid = threadIdx.x;
  const float* rp = in + (size_t)row * D;
  float sum = 0.f, sq = 0.f;
  for (int i = tid; i < D; i += blockDim.x) {
    float v = rp[i] + (addb ? addb[i] : 0.f);
    sum += v; sq += v * v;
  }
  s1[tid] = sum; s2[tid] = sq; __syncthreads();
  for (int off = 128; off; off >>= 1) {
    if (tid < off) { s1[tid] += s1[tid + off]; s2[tid] += s2[tid + off]; }
    __syncthreads();
  }
  float mu = s1[0] / (float)D;
  float var = s2[0] / (float)D - mu * mu;
  float inv = rsqrtf(var + 1e-5f);
  for (int i = tid; i < D; i += blockDim.x) {
    float v = rp[i] + (addb ? addb[i] : 0.f);
    out[(size_t)row * D + i] = (v - mu) * inv * g[i] + b[i];
  }
}

// =====================================================================
// Tail: attention fusion of [T,newT] -> W2 -> bn2 -> elu -> yce head
// =====================================================================
__global__ __launch_bounds__(256) void k_tail(const float* __restrict__ T, const float* __restrict__ nT,
                                              const float* __restrict__ attw,
                                              const float* __restrict__ W2, const float* __restrict__ b2,
                                              const float* __restrict__ g2, const float* __restrict__ bb2,
                                              const float* __restrict__ yw, const float* __restrict__ yb,
                                              float* out) {
  __shared__ float RH[IN_DIM];
  __shared__ float TEL[NCLS];
  __shared__ float red[256];
  __shared__ float sc[2];
  int tid = threadIdx.x;

  float a = 0.f;
  for (int j = tid; j < IN_DIM; j += 256) a += T[j] * attw[j];
  red[tid] = a; __syncthreads();
  for (int off = 128; off; off >>= 1) { if (tid < off) red[tid] += red[tid + off]; __syncthreads(); }
  if (tid == 0) sc[0] = red[0];
  __syncthreads();
  float b = 0.f;
  for (int j = tid; j < IN_DIM; j += 256) b += nT[j] * attw[j];
  red[tid] = b; __syncthreads();
  for (int off = 128; off; off >>= 1) { if (tid < off) red[tid] += red[tid + off]; __syncthreads(); }
  if (tid == 0) sc[1] = red[0];
  __syncthreads();

  float m = fmaxf(sc[0], sc[1]);
  float e0 = __expf(sc[0] - m), e1 = __expf(sc[1] - m);
  float inv = 1.f / (e0 + e1);
  float w0 = e0 * inv, w1 = e1 * inv;
  for (int j = tid; j < IN_DIM; j += 256) RH[j] = w0 * T[j] + w1 * nT[j];
  __syncthreads();

  for (int c = tid; c < NCLS; c += 256) {
    float t = b2[c];
    for (int j = 0; j < IN_DIM; ++j) t += RH[j] * W2[(size_t)j * NCLS + c];
    t = g2[c] * t * BN_SCALE + bb2[c];
    TEL[c] = elu1(t);
  }
  __syncthreads();

  float o0 = 0.f, o1 = 0.f;
  for (int c = tid; c < NCLS; c += 256) { o0 += TEL[c] * yw[c * 2 + 0]; o1 += TEL[c] * yw[c * 2 + 1]; }
  red[tid] = o0; __syncthreads();
  for (int off = 128; off; off >>= 1) { if (tid < off) red[tid] += red[tid + off]; __syncthreads(); }
  if (tid == 0) out[0] = red[0] + yb[0];
  __syncthreads();
  red[tid] = o1; __syncthreads();
  for (int off = 128; off; off >>= 1) { if (tid < off) red[tid] += red[tid + off]; __syncthreads(); }
  if (tid == 0) out[1] = red[0] + yb[1];
}

// =====================================================================
// Host orchestration
// =====================================================================
extern "C" void kernel_launch(void* const* d_in, const int* in_sizes, int n_in,
                              void* d_out, int out_size, void* d_ws, size_t ws_size,
                              hipStream_t stream) {
  (void)n_in; (void)out_size; (void)ws_size;

  // top-level order: insertion (h,edges0,edges1,params) or sorted (edges0,edges1,h,params)
  const float* h; const int* edges0; const int* edges1;
  const int pbase = 3;
  if (in_sizes[0] == N_NODES * IN_DIM) {
    h = (const float*)d_in[0]; edges0 = (const int*)d_in[1]; edges1 = (const int*)d_in[2];
  } else {
    edges0 = (const int*)d_in[0]; edges1 = (const int*)d_in[1]; h = (const float*)d_in[2];
  }
  auto P = [&](int i) { return (const float*)d_in[pbase + i]; };
  // params leaves, sorted-key DFS flatten order
  const float *W2 = P(0), *W3 = P(1), *W4 = P(2), *attw = P(3), *b2 = P(4), *b3 = P(5), *b4 = P(6);
  const float *bn1b = P(7), *bn1g = P(8), *bn2b = P(9), *bn2g = P(10), *bn3b = P(11), *bn3g = P(12);
  const float *Wk = P(13), *Wp = P(14), *Wq = P(15), *Wv = P(16), *bp = P(17);
  const float *lnb1 = P(18), *lnb2 = P(19), *lng1 = P(20), *lng2 = P(21);
  const float* gatP[2][2][4];   // [layer][metapath][W,al,ar,b]
  for (int l = 0; l < 2; ++l)
    for (int m = 0; m < 2; ++m)
      for (int j = 0; j < 4; ++j) gatP[l][m][j] = P(22 + (l * 2 + m) * 4 + j);
  const float *gate_b = P(38), *gate_w = P(39);
  const float *semW1[2] = { P(40), P(43) }, *semb1[2] = { P(41), P(44) }, *semw2[2] = { P(42), P(45) };
  const float *yce_b = P(46), *yce_w = P(47);

  // workspace layout (floats)
  float* ws = (float*)d_ws;
  const size_t NHDs = (size_t)N_NODES * HD;       // 10.24M
  const size_t NINs = (size_t)N_NODES * IN_DIM;   // 15.36M
  const size_t NH   = (size_t)N_NODES * HEADS;
  size_t o = 0;
  float* xbuf = ws + o; o += NHDs;
  float* h3   = ws + o; o += NHDs;
  float* z0   = ws + o; o += NHDs;
  float* z1   = ws + o; o += NHDs;
  float* fbuf = ws + o; o += NHDs;
  float* sem0 = ws + o; o += (size_t)N_NODES * 128;
  float* sem1 = ws + o; o += (size_t)N_NODES * 128;
  float* qbuf = z0;      // aliases z0+z1 (dead at fusion stage); 15.36M fits in 20.48M
  float* kbuf = fbuf;    // aliases fbuf+sem0+sem1 = exactly 15.36M
  float* Pa   = ws + o; o += NINs;   // hv -> b_ (in-place LN)
  float* Pb   = ws + o; o += NINs;   // a_
  float* ee   = ws + o; o += (size_t)N_EDGES * HEADS;
  float* elb  = ws + o; o += NH;
  float* erb  = ws + o; o += NH;
  float* ssb  = ws + o; o += NH;
  unsigned* mord = (unsigned*)(ws + o); o += NH;
  float* gate = ws + o; o += N_NODES;
  float* svec = ws + o; o += N_NODES;
  float* wsum = ws + o; o += 2;
  float* beta = ws + o; o += 2;
  float* Tp   = ws + o; o += HD;
  float* Tvec = ws + o; o += IN_DIM;
  float* vsum = ws + o; o += IN_DIM;
  float* nTv  = ws + o; o += IN_DIM;
  unsigned* rmax = (unsigned*)(ws + o); o += 1;
  float* rsum = ws + o; o += 1;

  auto gemm = [&](const float* A, const float* B, const float* bias, float* C, int M, int Ncol, int K) {
    dim3 grid((unsigned)(Ncol / BN), (unsigned)((M + BM - 1) / BM));
    gemm_bf16_wmma<<<grid, 256, 0, stream>>>(A, B, bias, C, M, Ncol, K);
  };
  auto fillf = [&](float* p, float v, long n) {
    k_fill_f32<<<dim3((unsigned)((n + 255) / 256)), 256, 0, stream>>>(p, v, n);
  };
  auto fillu = [&](unsigned* p, unsigned v, long n) {
    k_fill_u32<<<dim3((unsigned)((n + 255) / 256)), 256, 0, stream>>>(p, v, n);
  };

  // ---- h3 = h @ W3 + b3 (computed once from original h) ----
  gemm(h, W3, b3, h3, N_NODES, HD, IN_DIM);

  // ---- HAN layers ----
  for (int l = 0; l < 2; ++l) {
    const float* xin = (l == 0) ? h : xbuf;
    const int Kin = (l == 0) ? IN_DIM : HD;
    for (int m = 0; m < 2; ++m) {
      const int* edg = (m == 0) ? edges0 : edges1;
      const int* src = edg;
      const int* dst = edg + N_EDGES;
      float* z = m ? z1 : z0;

      gemm(xin, gatP[l][m][0], nullptr, fbuf, N_NODES, HD, Kin);   // f = x@W
      long nh = (long)N_NODES * HEADS;
      k_attn_logits<<<(unsigned)((nh + 255) / 256), 256, 0, stream>>>(
          fbuf, gatP[l][m][1], gatP[l][m][2], elb, erb, nh);
      fillu(mord, 0x007FFFFFu, nh);   // f2o(-inf)
      fillf(ssb, 0.f, nh);
      fillf(z, 0.f, (long)NHDs);
      long eh = (long)N_EDGES * HEADS;
      k_edge_max<<<(unsigned)((eh + 255) / 256), 256, 0, stream>>>(elb, erb, src, dst, ee, mord, eh);
      k_edge_expsum<<<(unsigned)((eh + 255) / 256), 256, 0, stream>>>(ee, dst, mord, ssb, eh);
      long ehd = (long)N_EDGES * HD;
      k_edge_agg<<<(unsigned)((ehd + 255) / 256), 256, 0, stream>>>(ee, ssb, fbuf, src, dst, z, ehd);
      k_bias_elu<<<(unsigned)((NHDs + 255) / 256), 256, 0, stream>>>(z, gatP[l][m][3], (long)NHDs);
    }
    // semantic attention
    gemm(z0, semW1[l], nullptr, sem0, N_NODES, 128, HD);
    gemm(z1, semW1[l], nullptr, sem1, N_NODES, 128, HD);
    fillf(wsum, 0.f, 2);
    k_sem_reduce<<<(N_NODES + 255) / 256, 256, 0, stream>>>(sem0, semb1[l], semw2[l], wsum + 0);
    k_sem_reduce<<<(N_NODES + 255) / 256, 256, 0, stream>>>(sem1, semb1[l], semw2[l], wsum + 1);
    k_beta<<<1, 1, 0, stream>>>(wsum, beta);
    k_combine<<<(unsigned)((NHDs + 255) / 256), 256, 0, stream>>>(h3, z0, z1, beta, bn1g, bn1b, xbuf, (long)NHDs);
  }

  // ---- global attention pooling -> T ----
  k_rowdot<<<2500, 256, 0, stream>>>(xbuf, gate_w, gate_b, gate, N_NODES, HD);
  fillu(rmax, 0x007FFFFFu, 1);
  fillf(rsum, 0.f, 1);
  k_reduce_max<<<80, 256, 0, stream>>>(gate, N_NODES, rmax);
  k_reduce_sumexp<<<80, 256, 0, stream>>>(gate, N_NODES, rmax, rsum);
  k_wcolsum<<<2, 256, 0, stream>>>(xbuf, gate, rmax, rsum, Tp, N_NODES, HD);
  k_vecmat<<<3, 256, 0, stream>>>(Tp, W4, b4, Tvec, HD, IN_DIM);

  // ---- LayerNorm dot-product attention fusion -> newT ----
  gemm(xbuf, Wp, nullptr, Pa, N_NODES, IN_DIM, HD);                        // hv (bp added in LN)
  k_layernorm<<<N_NODES, 256, 0, stream>>>(h, nullptr, lng1, lnb1, Pb, IN_DIM);   // a_
  k_layernorm<<<N_NODES, 256, 0, stream>>>(Pa, bp, lng2, lnb2, Pa, IN_DIM);       // b_ (in place)
  gemm(Pb, Wq, nullptr, qbuf, N_NODES, IN_DIM, IN_DIM);                    // q = a_@Wq
  gemm(Pa, Wk, nullptr, kbuf, N_NODES, IN_DIM, IN_DIM);                    // k = b_@Wk
  k_rowdot2<<<2500, 256, 0, stream>>>(qbuf, kbuf, svec, N_NODES, IN_DIM, 1.0f / sqrtf(768.f));
  fillu(rmax, 0x007FFFFFu, 1);
  fillf(rsum, 0.f, 1);
  k_reduce_max<<<80, 256, 0, stream>>>(svec, N_NODES, rmax);
  k_reduce_sumexp<<<80, 256, 0, stream>>>(svec, N_NODES, rmax, rsum);
  // newT = softmax(s) @ (b_ @ Wv) == (softmax(s) @ b_) @ Wv  (factored)
  k_wcolsum<<<3, 256, 0, stream>>>(Pa, svec, rmax, rsum, vsum, N_NODES, IN_DIM);
  k_newT<<<3, 256, 0, stream>>>(vsum, Wv, bn3g, bn3b, nTv, IN_DIM, IN_DIM);

  // ---- final fusion + classifier head ----
  k_tail<<<1, 256, 0, stream>>>(Tvec, nTv, attw, W2, b2, bn2g, bn2b, yce_w, yce_b, (float*)d_out);
}
